// JointNetwork_59871844106969
// MI455X (gfx1250) — compile-verified, matrix operand order checked by reading, not run on hardware
//
#include <hip/hip_runtime.h>

// RNN-T Joint Network for MI455X (gfx1250), wave32 + WMMA f16 path.
// B=8, T=256, U=64, D_ENC=D_DEC=512, J=640, V=1024.
//
// Round 5: fully unroll the K loops (K is compile-time constant). The manual
// double-buffer forced cross-backedge fixed registers -> v_dual_mov rotation
// copies + WMMA->VALU hazard v_nops. Full unroll makes all fragments SSA so
// loads allocate straight into contiguous WMMA operand groups and the
// pressure-aware scheduler does the pipelining.

typedef __attribute__((ext_vector_type(16))) _Float16 v16h;
typedef __attribute__((ext_vector_type(8)))  float    v8f;

union Frag {
    v16h  h;
    uint4 q[2];
};

// Load one 16x32 f16 WMMA operand from row-major-K storage.
// ISA 16-bit operand layout (CDNA5 ISA 7.12.2):
//   lanes 0-15 : row/col = lane,    K = {k0..k0+7, k0+16..k0+23}
//   lanes 16-31: row/col = lane-16, K = {k0+8..k0+15, k0+24..k0+31}
__device__ __forceinline__ v16h load_frag(const _Float16* __restrict__ base,
                                          int strideH, int k0, int lane) {
    const int hi = (lane >> 4) & 1;
    const _Float16* p = base + (lane & 15) * strideH + k0 + hi * 8;
    Frag f;
    f.q[0] = *(const uint4*)(p);        // 8 halves (16B)
    f.q[1] = *(const uint4*)(p + 16);   // 8 halves (16B)
    return f.h;
}

#define JDIM 640
#define LDS_STRIDE 648   // +8 halves pad: conflict-free row-parallel ds_load_b128

// ---------------------------------------------------------------------------
// Stage 0: f32 -> f16 conversion (vectorized x4)
// ---------------------------------------------------------------------------
__global__ void f32_to_f16_v4(const float* __restrict__ s,
                              _Float16* __restrict__ d, int n4) {
    int i = blockIdx.x * blockDim.x + threadIdx.x;
    if (i >= n4) return;
    float4 v = ((const float4*)s)[i];
    union { _Float16 h[4]; uint2 u; } pk;
    pk.h[0] = (_Float16)v.x;
    pk.h[1] = (_Float16)v.y;
    pk.h[2] = (_Float16)v.z;
    pk.h[3] = (_Float16)v.w;
    ((uint2*)d)[i] = pk.u;
}

// ---------------------------------------------------------------------------
// Stage 1: projection GEMM  Y(M,640) = X(M,512) @ W(640,512)^T + bias
// 8 waves/block, each wave owns 16 rows; N blocked 4 tiles; K fully unrolled.
// ---------------------------------------------------------------------------
__global__ __launch_bounds__(256)
void proj_gemm_f16(const _Float16* __restrict__ X,
                   const _Float16* __restrict__ W,
                   const float* __restrict__ bias,
                   float* __restrict__ Y) {
    const int lane    = threadIdx.x & 31;
    const int wave    = threadIdx.x >> 5;
    const int rowBase = blockIdx.x * 128 + wave * 16;
    const _Float16* Xrow = X + (size_t)rowBase * 512;
    const int nCol = lane & 15;
    const int mOff = (lane < 16) ? 0 : 8;

    #pragma unroll 1
    for (int ng = 0; ng < 10; ++ng) {          // 10 groups * 64 cols = 640
        const int n0 = ng * 64;
        v8f acc[4];
        #pragma unroll
        for (int nt = 0; nt < 4; ++nt) {
            const float bv = bias[n0 + nt * 16 + nCol];
            v8f ini = {bv, bv, bv, bv, bv, bv, bv, bv};
            acc[nt] = ini;
        }
        #pragma unroll
        for (int k0 = 0; k0 < 512; k0 += 32) {             // 16 steps, SSA
            v16h a = load_frag(Xrow, 512, k0, lane);
            #pragma unroll
            for (int nt = 0; nt < 4; ++nt) {
                v16h b = load_frag(W + (size_t)(n0 + nt * 16) * 512,
                                   512, k0, lane);
                acc[nt] = __builtin_amdgcn_wmma_f32_16x16x32_f16(
                    false, a, false, b, (short)0, acc[nt], false, false);
            }
        }
        // C layout: VGPR v -> M = v (lanes 0-15) / v+8 (lanes 16-31)
        #pragma unroll
        for (int nt = 0; nt < 4; ++nt) {
            float* yr = Y + (size_t)(rowBase + mOff) * 640 + (n0 + nt * 16 + nCol);
            #pragma unroll
            for (int v = 0; v < 8; ++v)
                yr[(size_t)v * 640] = acc[nt][v];
        }
    }
}

// ---------------------------------------------------------------------------
// Stage 2: fused tanh + big GEMM
// out(131072,1024) = tanh(encP[bt] + decP[bu]) @ Wout(1024,640)^T + b_out
// Block = 128 joint rows x all 1024 cols. tanh'ed f16 A-tile lives in LDS,
// computed exactly once, reused for all N. Waves arranged 2(M) x 4(N), each
// wave computes a 64x64 output block (4x4 WMMA tiles) per pass; 4 passes.
// K loop fully unrolled (20 steps).
// ---------------------------------------------------------------------------
__global__ __launch_bounds__(256)
void joint_gemm(const float* __restrict__ encP,   // (2048, 640) f32
                const float* __restrict__ decP,   // (512, 640)  f32
                const _Float16* __restrict__ Wo,  // (1024, 640) f16
                const float* __restrict__ bo,     // (1024)      f32
                float* __restrict__ out) {        // (131072, 1024) f32
    __shared__ _Float16 As[128 * LDS_STRIDE];     // ~162 KB of 320 KB WGP LDS
    const int t     = threadIdx.x;
    const int mBase = blockIdx.x * 128;

    // ---- Phase 1: A = (f16) tanh(encP + decP), 4 elements/thread/iter ----
    // 128*640 = 81920 halves = 256 threads * 80 iters * 4
    #pragma unroll 1
    for (int i = 0; i < 80; ++i) {
        const int e   = (i * 256 + t) * 4;
        const int row = e / JDIM;
        const int k   = e % JDIM;
        const int r       = mBase + row;
        const int encRow  = r >> 6;                        // b*T + t  (r / U)
        const int decRow  = ((r >> 14) << 6) | (r & 63);   // b*U + u
        float4 a4 = *(const float4*)(encP + (size_t)encRow * JDIM + k);
        float4 b4 = *(const float4*)(decP + (size_t)decRow * JDIM + k);
        union { _Float16 h[4]; uint2 u; } pk;
        pk.h[0] = (_Float16)tanhf(a4.x + b4.x);
        pk.h[1] = (_Float16)tanhf(a4.y + b4.y);
        pk.h[2] = (_Float16)tanhf(a4.z + b4.z);
        pk.h[3] = (_Float16)tanhf(a4.w + b4.w);
        *(uint2*)&As[row * LDS_STRIDE + k] = pk.u;         // ds_store_b64
    }
    __syncthreads();

    // ---- Phase 2: waves 2(M) x 4(N); per pass each wave does 64x64 ----
    const int lane  = t & 31;
    const int wave  = t >> 5;
    const int waveM = wave & 1;        // 0..1 -> row block 64*waveM
    const int waveN = wave >> 1;       // 0..3 -> col block 64*waveN
    const _Float16* Abase = &As[(waveM * 64) * LDS_STRIDE];
    const int nCol = lane & 15;
    const int mOff = (lane < 16) ? 0 : 8;

    #pragma unroll 1
    for (int p = 0; p < 4; ++p) {
        const int nB = p * 256 + waveN * 64;
        v8f acc[4][4];
        #pragma unroll
        for (int nt = 0; nt < 4; ++nt) {
            const float bv = bo[nB + nt * 16 + nCol];
            v8f ini = {bv, bv, bv, bv, bv, bv, bv, bv};
            #pragma unroll
            for (int mt = 0; mt < 4; ++mt)
                acc[mt][nt] = ini;
        }
        #pragma unroll
        for (int k0 = 0; k0 < JDIM; k0 += 32) {            // 20 steps, SSA
            v16h a[4];
            #pragma unroll
            for (int mt = 0; mt < 4; ++mt)                 // ds_load_b128 x2 each
                a[mt] = load_frag(Abase + mt * 16 * LDS_STRIDE,
                                  LDS_STRIDE, k0, lane);
            #pragma unroll
            for (int nt = 0; nt < 4; ++nt) {
                v16h b = load_frag(Wo + (size_t)(nB + nt * 16) * JDIM,
                                   JDIM, k0, lane);        // global b128 x2 (L2-hot)
                #pragma unroll
                for (int mt = 0; mt < 4; ++mt)
                    acc[mt][nt] = __builtin_amdgcn_wmma_f32_16x16x32_f16(
                        false, a[mt], false, b, (short)0, acc[mt][nt],
                        false, false);
            }
        }
        // Streaming epilogue: non-temporal stores keep Wo/decP hot in L2.
        #pragma unroll
        for (int mt = 0; mt < 4; ++mt) {
            const int rowTop = mBase + waveM * 64 + mt * 16 + mOff;
            #pragma unroll
            for (int nt = 0; nt < 4; ++nt) {
                const int n = nB + nt * 16 + nCol;
                float* orow = out + (size_t)rowTop * 1024 + n;
                #pragma unroll
                for (int v = 0; v < 8; ++v)
                    __builtin_nontemporal_store(acc[mt][nt][v],
                                                orow + (size_t)v * 1024);
            }
        }
    }
}

// ---------------------------------------------------------------------------
// Host launcher
// ---------------------------------------------------------------------------
extern "C" void kernel_launch(void* const* d_in, const int* in_sizes, int n_in,
                              void* d_out, int out_size, void* d_ws, size_t ws_size,
                              hipStream_t stream) {
    const float* enc_out = (const float*)d_in[0];   // 8*256*512  = 1048576
    const float* dec_out = (const float*)d_in[1];   // 8*64*512   = 262144
    const float* W_enc   = (const float*)d_in[2];   // 640*512    = 327680
    const float* b_enc   = (const float*)d_in[3];   // 640
    const float* W_dec   = (const float*)d_in[4];   // 640*512    = 327680
    const float* b_dec   = (const float*)d_in[5];   // 640
    const float* W_out   = (const float*)d_in[6];   // 1024*640   = 655360
    const float* b_out   = (const float*)d_in[7];   // 1024
    float* out = (float*)d_out;                     // 131072*1024

    // Workspace layout (bytes), ~11.8 MB total
    char* ws = (char*)d_ws;
    float*     encP   = (float*)(ws + 0);           // 2048*640*4 = 5242880
    float*     decP   = (float*)(ws + 5242880);     //  512*640*4 = 1310720
    _Float16*  enc_h  = (_Float16*)(ws + 6553600);  // 1048576*2  = 2097152
    _Float16*  dec_h  = (_Float16*)(ws + 8650752);  //  262144*2  =  524288
    _Float16*  Wenc_h = (_Float16*)(ws + 9175040);  //  327680*2  =  655360
    _Float16*  Wdec_h = (_Float16*)(ws + 9830400);  //  327680*2  =  655360
    _Float16*  Wout_h = (_Float16*)(ws + 10485760); //  655360*2  = 1310720

    // Stage 0: f32 -> f16
    auto cvt = [&](const float* s, _Float16* d, int n) {
        const int n4 = n / 4;
        f32_to_f16_v4<<<(n4 + 255) / 256, 256, 0, stream>>>(s, d, n4);
    };
    cvt(enc_out, enc_h, 1048576);
    cvt(dec_out, dec_h, 262144);
    cvt(W_enc,   Wenc_h, 327680);
    cvt(W_dec,   Wdec_h, 327680);
    cvt(W_out,   Wout_h, 655360);

    // Stage 1: projections (f32 accumulate, bias fused)
    proj_gemm_f16<<<16, 256, 0, stream>>>(enc_h, Wenc_h, b_enc, encP); // M=2048
    proj_gemm_f16<<< 4, 256, 0, stream>>>(dec_h, Wdec_h, b_dec, decP); // M=512

    // Stage 2: fused tanh + (131072 x 1024 x 640) GEMM
    joint_gemm<<<1024, 256, 0, stream>>>(encP, decP, Wout_h, b_out, out);
}
